// Net_90426241450695
// MI455X (gfx1250) — matrix operand order repaired and checked by reading, not compiled, not run on hardware
//
#include <hip/hip_runtime.h>

typedef _Float16 v16h __attribute__((ext_vector_type(16)));
typedef _Float16 v8h  __attribute__((ext_vector_type(8)));
typedef float    v8f  __attribute__((ext_vector_type(8)));
typedef unsigned int v4u __attribute__((ext_vector_type(4)));
typedef int      v8i  __attribute__((ext_vector_type(8)));
typedef int      v4i  __attribute__((ext_vector_type(4)));

#define IMG 4096
#define OUTD 2048
#define IN_TILE 144          // 64x64 final tile needs 144x144 input (halo 8)
#define STRIDE_BIG 152       // halves; 304B row stride, 16B aligned
#define ROWS_BIG 148
#define STRIDE_SM 88         // halves; 176B row stride, 16B aligned
#define ROWS_SM 88

#define SZ_BIG (ROWS_BIG * STRIDE_BIG * 2)   // 44992 B
#define SZ_SM  (ROWS_SM * STRIDE_SM * 2)     // 15488 B
#define OFF_A 0
#define OFF_B (OFF_A + SZ_BIG)
#define OFF_P (OFF_B + SZ_BIG)
#define OFF_3 (OFF_P + SZ_SM)
#define OFF_4 (OFF_3 + SZ_SM)
#define OFF_S (OFF_4 + SZ_SM)                // 136448 B
#define SZ_S  (IN_TILE * IN_TILE * 4)        // 82944 B (fp32 staging for TDM)
#define SMEM_BYTES (OFF_S + SZ_S)            // 219392 B < 320KB

#define NTHREADS 512
#define NWAVES 16

// --- Banded weight operand: B[k][n] = w[dy][k-n] for k-n in {0,1,2}, K>=18 zero.
// Dense 16-bit B 32x16 layout (matching A): lanes 0-15 hold K=0..7 | K16..23 (col n = lane),
// lanes 16-31 hold K=8..15 | K24..31 (col n = lane-16).
struct B3 { v16h b[3]; };

__device__ __forceinline__ B3 make_B(const float* __restrict__ w, int lane) {
  float wf[9];
#pragma unroll
  for (int i = 0; i < 9; ++i) wf[i] = w[i];
  int n  = lane & 15;
  int kb = (lane >> 4) << 3;
  union { v16h v; _Float16 e[16]; } u[3];
#pragma unroll
  for (int dy = 0; dy < 3; ++dy)
#pragma unroll
    for (int j = 0; j < 16; ++j) {
      int d = kb + (j & 7) + ((j >> 3) << 4) - n;  // k - n ; k <= 17 when d in [0,2]
      u[dy].e[j] = (d >= 0 && d < 3) ? (_Float16)wf[dy * 3 + d] : (_Float16)0.0f;
    }
  B3 r; r.b[0] = u[0].v; r.b[1] = u[1].v; r.b[2] = u[2].v;
  return r;
}

// --- A operand: 16-bit A 16x32 layout. Lane<16: M=lane, halves = K0..7 | K16..23.
// Lane>=16: M=lane-16, halves = K8..15 | K24..31.  K index k maps input col ox+k.
__device__ __forceinline__ v16h pack_a(const _Float16* in, int istride,
                                       int oy, int ox, int dy, int lane) {
  int m   = lane & 15;
  int hi  = lane >> 4;
  int row = oy + dy + m;
  int col = ox + (hi << 3);
  const v8h* p = (const v8h*)(in + row * istride + col);
  v8h c0 = p[0];        // 8 halves at col
  v8h c1 = p[2];        // 8 halves at col+16
  return __builtin_shufflevector(c0, c1, 0,1,2,3,4,5,6,7,8,9,10,11,12,13,14,15);
}

__device__ __forceinline__ v8f conv_tile_acc(const _Float16* in, int istride,
                                             int oy, int ox, const B3& wb, int lane) {
  v8f acc = {0.f,0.f,0.f,0.f,0.f,0.f,0.f,0.f};
#pragma unroll
  for (int dy = 0; dy < 3; ++dy) {
    v16h a = pack_a(in, istride, oy, ox, dy, lane);
    acc = __builtin_amdgcn_wmma_f32_16x16x32_f16(false, a, false, wb.b[dy],
                                                 (short)0, acc, false, false);
  }
  return acc;
}

// One conv stage over a tile grid; stores f16 with zero-masking outside the
// feature map's global bounds (reproduces SAME zero-padding for the next stage).
__device__ __forceinline__ void conv_stage(const _Float16* in, int istride,
                                           _Float16* out, int ostride,
                                           int nt_row, int nt_total,
                                           const float* __restrict__ w,
                                           int gby, int gbx, int glim, int tid) {
  int lane = tid & 31, wid = tid >> 5;
  B3 wb = make_B(w, lane);
  for (int t = wid; t < nt_total; t += NWAVES) {
    int oy = (t / nt_row) << 4;
    int ox = (t % nt_row) << 4;
    v8f acc = conv_tile_acc(in, istride, oy, ox, wb, lane);
    // C/D layout: lane<16 -> N=lane, rows oy..oy+7 ; lane>=16 -> rows oy+8..oy+15
    int col = ox + (lane & 15);
    int gx  = gbx + col;
    bool xok = (gx >= 0) & (gx < glim);
    int rb = oy + ((lane >> 4) << 3);
#pragma unroll
    for (int r = 0; r < 8; ++r) {
      int row = rb + r;
      int gy  = gby + row;
      bool ok = xok & (gy >= 0) & (gy < glim);
      out[row * ostride + col] = ok ? (_Float16)acc[r] : (_Float16)0.0f;
    }
  }
}

extern "C" __global__ void __launch_bounds__(NTHREADS, 1)
fused_conv_pipeline(const float* __restrict__ x,
                    const float* __restrict__ w1, const float* __restrict__ w2,
                    const float* __restrict__ w3, const float* __restrict__ w4,
                    const float* __restrict__ w5, float* __restrict__ out) {
  extern __shared__ char smem[];
  _Float16* bufA = (_Float16*)(smem + OFF_A);
  _Float16* bufB = (_Float16*)(smem + OFF_B);
  _Float16* bufP = (_Float16*)(smem + OFF_P);
  _Float16* buf3 = (_Float16*)(smem + OFF_3);
  _Float16* buf4 = (_Float16*)(smem + OFF_4);
  const float* staging = (const float*)(smem + OFF_S);

  int tid = threadIdx.x;
  int tx = blockIdx.x, ty = blockIdx.y, img = blockIdx.z;

  // Input tile [gy0, gy0+144) x [gx0, gx0+144), clipped at image edges.
  int gy0 = ty * 128 - 8, gx0 = tx * 128 - 8;
  int gys = gy0 < 0 ? 0 : gy0;
  int gxs = gx0 < 0 ? 0 : gx0;
  int gye = gy0 + IN_TILE; if (gye > IMG) gye = IMG;
  int gxe = gx0 + IN_TILE; if (gxe > IMG) gxe = IMG;
  int hcl = gye - gys, wcl = gxe - gxs;
  int dy0 = gys - gy0, dx0 = gxs - gx0;

  // ---- TDM: async DMA of the fp32 tile (clipped) into LDS staging (wave 0 only;
  // EXEC is ignored by TENSOR_LOAD_TO_LDS, so one wave issues exactly one DMA).
  if (tid < 32) {
    unsigned long long ga = (unsigned long long)(const char*)x +
        (((unsigned long long)img * IMG + (unsigned)gys) * IMG + (unsigned)gxs) * 4ull;
    v4u g0;
    g0.x = 1u;                                        // count=1, user descriptor
    g0.y = (unsigned)OFF_S;                           // lds_addr (bytes)
    g0.z = (unsigned)(ga & 0xffffffffull);            // global_addr[31:0]
    g0.w = (unsigned)((ga >> 32) & 0x1ffffffull) | 0x80000000u;  // addr[56:32] | type=2
    v8i g1;
    g1[0] = (int)(2u << 16);                          // data_size = 4 bytes
    g1[1] = (int)((unsigned)IMG << 16);               // tensor_dim0[15:0] @ [63:48]
    g1[2] = (int)(((unsigned)IMG >> 16) | ((unsigned)IMG << 16));  // dim0 hi | dim1 lo
    g1[3] = (int)(((unsigned)IMG >> 16) | ((unsigned)wcl << 16));  // dim1 hi | tile_dim0
    g1[4] = (int)(unsigned)hcl;                       // tile_dim1 (tile_dim2 = 0 -> 2D)
    g1[5] = IMG;                                      // tensor_dim0_stride lo32
    g1[6] = 0;                                        // stride0 hi16 | stride1 lo16
    g1[7] = (int)(((unsigned long long)IMG * IMG) >> 16);  // stride1[47:16]
    v4i g2 = {0, 0, 0, 0}, g3 = {0, 0, 0, 0};
#if __clang_major__ >= 23
    v8i gz = {0, 0, 0, 0, 0, 0, 0, 0};
    __builtin_amdgcn_tensor_load_to_lds(g0, g1, g2, g3, gz, 0);
#else
    __builtin_amdgcn_tensor_load_to_lds(g0, g1, g2, g3, 0);
#endif
  }

  // ---- Zero all f16 working buffers while the DMA is in flight (disjoint region).
  {
    v4u z = {0, 0, 0, 0};
    v4u* s = (v4u*)smem;
    for (int i = tid; i < OFF_S / 16; i += NTHREADS) s[i] = z;
  }
  __syncthreads();
  if (tid < 32) __builtin_amdgcn_s_wait_tensorcnt(0);
  __syncthreads();

  // ---- fp32 -> fp16 with explicit zero halo (matches SAME zero padding).
  for (int i = tid; i < IN_TILE * IN_TILE; i += NTHREADS) {
    int r = i / IN_TILE, c = i % IN_TILE;
    float v = 0.0f;
    if (r >= dy0 && r < dy0 + hcl && c >= dx0 && c < dx0 + wcl)
      v = staging[(r - dy0) * wcl + (c - dx0)];
    bufA[r * STRIDE_BIG + c] = (_Float16)v;
  }
  __syncthreads();

  // conv1: 9x9 tiles of 16x16 over 144x144 (valid 142x142), local row r -> global gy0+r+1
  conv_stage(bufA, STRIDE_BIG, bufB, STRIDE_BIG, 9, 81, w1, gy0 + 1, gx0 + 1, IMG, tid);
  __syncthreads();
  // conv2: local row r -> global gy0+r+2 (valid 140x140)
  conv_stage(bufB, STRIDE_BIG, bufA, STRIDE_BIG, 9, 81, w2, gy0 + 2, gx0 + 2, IMG, tid);
  __syncthreads();

  // ---- 2x2 max pool: pool local p -> pool_global ty*64-3+p, reads conv2 rows 2p,2p+1
  int py0 = ty * 64 - 3, px0 = tx * 64 - 3;
  for (int i = tid; i < 70 * 70; i += NTHREADS) {
    int p = i / 70, q = i % 70;
    const _Float16* r0 = bufA + (2 * p) * STRIDE_BIG + 2 * q;
    const _Float16* r1 = r0 + STRIDE_BIG;
    float m = fmaxf(fmaxf((float)r0[0], (float)r0[1]),
                    fmaxf((float)r1[0], (float)r1[1]));
    int pg = py0 + p, qg = px0 + q;
    bool ok = (pg >= 0) & (pg < OUTD) & (qg >= 0) & (qg < OUTD);
    bufP[p * STRIDE_SM + q] = ok ? (_Float16)m : (_Float16)0.0f;
  }
  __syncthreads();

  // conv3 / conv4 on 5x5 tile grids over 80x80 (valid 68 / 66)
  conv_stage(bufP, STRIDE_SM, buf3, STRIDE_SM, 5, 25, w3, ty * 64 - 2, tx * 64 - 2, OUTD, tid);
  __syncthreads();
  conv_stage(buf3, STRIDE_SM, buf4, STRIDE_SM, 5, 25, w4, ty * 64 - 1, tx * 64 - 1, OUTD, tid);
  __syncthreads();

  // ---- conv5: 4x4 tiles, f32 accumulators straight to global output (always in-bounds)
  {
    int lane = tid & 31, wid = tid >> 5;
    B3 wb = make_B(w5, lane);
    for (int t = wid; t < 16; t += NWAVES) {
      int oy = (t >> 2) << 4, ox = (t & 3) << 4;
      v8f acc = conv_tile_acc(buf4, STRIDE_SM, oy, ox, wb, lane);
      int col = ox + (lane & 15);
      int rb  = oy + ((lane >> 4) << 3);
      size_t base = ((size_t)img * OUTD + (size_t)(ty * 64 + rb)) * OUTD
                    + (size_t)(tx * 64 + col);
#pragma unroll
      for (int r = 0; r < 8; ++r) out[base + (size_t)r * OUTD] = acc[r];
    }
  }
}

extern "C" void kernel_launch(void* const* d_in, const int* in_sizes, int n_in,
                              void* d_out, int out_size, void* d_ws, size_t ws_size,
                              hipStream_t stream) {
  const float* x  = (const float*)d_in[0];
  const float* w1 = (const float*)d_in[1];
  const float* w2 = (const float*)d_in[2];
  const float* w3 = (const float*)d_in[3];
  const float* w4 = (const float*)d_in[4];
  const float* w5 = (const float*)d_in[5];
  float* out = (float*)d_out;
  (void)in_sizes; (void)n_in; (void)out_size; (void)d_ws; (void)ws_size;

  (void)hipFuncSetAttribute((const void*)fused_conv_pipeline,
                            hipFuncAttributeMaxDynamicSharedMemorySize, SMEM_BYTES);
  // 2048/64 = 32 tiles per axis, 4 images
  fused_conv_pipeline<<<dim3(32, 32, 4), dim3(NTHREADS, 1, 1), SMEM_BYTES, stream>>>(
      x, w1, w2, w3, w4, w5, out);
}
